// EdgeAugmentedSelfAttention_60241211294243
// MI455X (gfx1250) — matrix operand order repaired and checked by reading, not compiled, not run on hardware
//
#include <hip/hip_runtime.h>
#include <hip/hip_bf16.h>

typedef __attribute__((ext_vector_type(16))) __bf16 v16bf;
typedef __attribute__((ext_vector_type(8)))  float  v8f;
typedef __attribute__((ext_vector_type(4)))  unsigned int u32x4;
typedef __attribute__((ext_vector_type(4)))  float  f32x4;

#define HEADS   8
#define NODE_WD 512
#define EDGE_WD 32
#define DOTD    64
#define BSZ     4
#define NN      1024

union FragB16 { v16bf v; u32x4 u[2]; };

__device__ __forceinline__ v8f wmma_bf16(v16bf a, v16bf b, v8f c) {
  return __builtin_amdgcn_wmma_f32_16x16x32_bf16(false, a, false, b, (short)0, c, false, false);
}

__device__ __forceinline__ v8f v8f_zero() {
  v8f z;
#pragma unroll
  for (int i = 0; i < 8; ++i) z[i] = 0.0f;
  return z;
}

// ---------------------------------------------------------------------------
// Kernel 0: precision / layout prep.
//  nodes_bf[b][n][k]      = bf16(nodes)
//  Wt[w][col][k]          = bf16(W_w[k][col])        (w: 0=q,1=k,2=v)
//  Wcombt[col][k] (16x32) = col<8 ? We[k][col] : Wg[k][col-8]   (bf16)
// ---------------------------------------------------------------------------
__global__ void prep_kernel(const float* __restrict__ nodes,
                            const float* __restrict__ Wq, const float* __restrict__ Wk,
                            const float* __restrict__ Wv, const float* __restrict__ Wg,
                            const float* __restrict__ We,
                            __bf16* __restrict__ nodes_bf, __bf16* __restrict__ Wt,
                            __bf16* __restrict__ Wcombt) {
  int i = blockIdx.x * 256 + threadIdx.x;
  const int NNODES = BSZ * NN * NODE_WD;           // 2,097,152
  if (i < NNODES) nodes_bf[i] = (__bf16)nodes[i];
  const int NW = NODE_WD * NODE_WD;                // 262,144
  if (i < 3 * NW) {
    int w = i / NW, r = i % NW;
    int col = r / NODE_WD, k = r % NODE_WD;
    const float* W = (w == 0) ? Wq : ((w == 1) ? Wk : Wv);
    Wt[i] = (__bf16)W[k * NODE_WD + col];
  }
  if (i < 16 * EDGE_WD) {
    int col = i / EDGE_WD, k = i % EDGE_WD;
    float v = (col < 8) ? We[k * HEADS + col] : Wg[k * HEADS + (col - 8)];
    Wcombt[i] = (__bf16)v;
  }
}

// ---------------------------------------------------------------------------
// Kernel 1: QKV projections via WMMA bf16 (f32 accumulate).
// One wave task = (weight w, 16-row tile, 64-col strip). 6144 tasks total.
// Outputs (bf16, head-major):
//   Qb,Kb: [b][h][n][64]     Vtb: [b][h][64][n]   (transposed for AV B-frags)
// ---------------------------------------------------------------------------
__global__ void qkv_kernel(const __bf16* __restrict__ nodes_bf, const __bf16* __restrict__ Wt,
                           const float* __restrict__ bq, const float* __restrict__ bk,
                           const float* __restrict__ bv,
                           __bf16* __restrict__ Qb, __bf16* __restrict__ Kb,
                           __bf16* __restrict__ Vtb) {
  const int lane = threadIdx.x & 31;
  const int wave = threadIdx.x >> 5;
  const int task = blockIdx.x * 8 + wave;          // 0..6143
  const int w    = task / 2048;                    // weight: 0=q,1=k,2=v
  const int rem  = task % 2048;
  const int rt   = rem / 8;                        // 256 row tiles
  const int cs   = rem % 8;                        // 8 col strips of 64
  const int b    = rt / 64;
  const int n0   = (rt % 64) * 16;
  const int c0   = cs * 64;
  const int hl   = lane >> 4;
  const int lm   = lane & 15;

  const __bf16* nrow = nodes_bf + ((size_t)(b * NN + n0 + lm)) * NODE_WD;
  v8f acc[4];
#pragma unroll
  for (int ct = 0; ct < 4; ++ct) acc[ct] = v8f_zero();

  for (int kc = 0; kc < 16; ++kc) {
    FragB16 A;                                     // A layout: runs {8hl..+7},{16+8hl..+7}
    A.u[0] = *(const u32x4*)(nrow + kc * 32 + 8 * hl);
    A.u[1] = *(const u32x4*)(nrow + kc * 32 + 16 + 8 * hl);
#pragma unroll
    for (int ct = 0; ct < 4; ++ct) {
      const __bf16* wrow = Wt + ((size_t)w) * NODE_WD * NODE_WD +
                           (size_t)(c0 + ct * 16 + lm) * NODE_WD + kc * 32;
      FragB16 Bf;                                  // B layout: runs {8hl..+7},{16+8hl..+7}
      Bf.u[0] = *(const u32x4*)(wrow + 8 * hl);
      Bf.u[1] = *(const u32x4*)(wrow + 16 + 8 * hl);
      acc[ct] = wmma_bf16(A.v, Bf.v, acc[ct]);
    }
  }

  const float* bias = (w == 0) ? bq : ((w == 1) ? bk : bv);
#pragma unroll
  for (int ct = 0; ct < 4; ++ct) {
    int c = c0 + ct * 16 + lm;
    float bval = bias[c];
    int h = c & 7, d = c >> 3;                     // reshape(.., DOT, HEADS): c = d*8 + h
#pragma unroll
    for (int r = 0; r < 8; ++r) {
      int n = n0 + r + 8 * hl;
      __bf16 vb = (__bf16)(acc[ct][r] + bval);
      if (w == 0)      Qb [(((size_t)b * HEADS + h) * NN + n) * DOTD + d] = vb;
      else if (w == 1) Kb [(((size_t)b * HEADS + h) * NN + n) * DOTD + d] = vb;
      else             Vtb[(((size_t)b * HEADS + h) * DOTD + d) * NN + n] = vb;
    }
  }
}

// ---------------------------------------------------------------------------
// Kernel 2: fused edge-augmented attention (flash-style online softmax).
// Grid: 256 WGs = (b, 16-row l tile). 8 waves = 8 heads.
// Streams edges exactly once; writes H (f32) and attn (f32).
// ---------------------------------------------------------------------------
__global__ void attn_kernel(const float* __restrict__ edges,
                            const __bf16* __restrict__ Qb, const __bf16* __restrict__ Kb,
                            const __bf16* __restrict__ Vtb, const __bf16* __restrict__ Wcombt,
                            const float* __restrict__ bg, const float* __restrict__ be,
                            float* __restrict__ attn_out, float* __restrict__ H_out) {
  __shared__ __align__(16) float  eg[16 * 32 * 16];     // [l][m][0..7=e, 8..15=g]  32KB
  __shared__ __align__(16) __bf16 Pst[8][16 * 32];      // per-wave gated probs      8KB

  const int lane = threadIdx.x & 31;
  const int wave = threadIdx.x >> 5;                    // == head h
  const int h    = wave;
  const int b    = blockIdx.x >> 6;
  const int l0   = (blockIdx.x & 63) * 16;
  const int hl   = lane >> 4;
  const int lm   = lane & 15;

  // Q A-frags (held in registers for the whole strip)
  FragB16 QA[2];
  {
    const __bf16* qrow = Qb + (((size_t)b * HEADS + h) * NN + l0 + lm) * DOTD;
    QA[0].u[0] = *(const u32x4*)(qrow + 8 * hl);
    QA[0].u[1] = *(const u32x4*)(qrow + 16 + 8 * hl);
    QA[1].u[0] = *(const u32x4*)(qrow + 32 + 8 * hl);
    QA[1].u[1] = *(const u32x4*)(qrow + 48 + 8 * hl);
  }
  // Combined [We|Wg] B-frag (32 x 16)
  FragB16 WB;
  {
    const __bf16* wrow = Wcombt + lm * EDGE_WD;
    WB.u[0] = *(const u32x4*)(wrow + 8 * hl);
    WB.u[1] = *(const u32x4*)(wrow + 16 + 8 * hl);
  }
  const float bias_lane = (lm < 8) ? be[lm] : bg[lm - 8];

  float mrun[8], lrun[8];
  v8f O[4];
#pragma unroll
  for (int r = 0; r < 8; ++r) { mrun[r] = -1e30f; lrun[r] = 0.0f; }
#pragma unroll
  for (int ct = 0; ct < 4; ++ct) O[ct] = v8f_zero();

  for (int mc = 0; mc < 32; ++mc) {
    const int m0 = mc * 32;

    // ---- (a) e/g projection for the 16l x 32m tile via WMMA ----
#pragma unroll
    for (int gi = 0; gi < 4; ++gi) {
      int grp   = wave * 4 + gi;                        // 32 groups of 16 (l,m) pairs
      int l_loc = grp >> 1;
      int mbase = (grp & 1) * 16;
      const float* erow = edges +
          ((((size_t)b * NN + l0 + l_loc) * NN) + m0 + mbase + lm) * EDGE_WD;
      __builtin_prefetch(erow + 32 * EDGE_WD, 0, 0);    // next m-chunk
      f32x4 e0 = *(const f32x4*)(erow + 8 * hl);
      f32x4 e1 = *(const f32x4*)(erow + 8 * hl + 4);
      f32x4 e2 = *(const f32x4*)(erow + 16 + 8 * hl);
      f32x4 e3 = *(const f32x4*)(erow + 16 + 8 * hl + 4);
      v16bf av;
#pragma unroll
      for (int t = 0; t < 4; ++t) {
        av[t]      = (__bf16)e0[t];
        av[4 + t]  = (__bf16)e1[t];
        av[8 + t]  = (__bf16)e2[t];
        av[12 + t] = (__bf16)e3[t];
      }
      v8f D = wmma_bf16(av, WB.v, v8f_zero());
#pragma unroll
      for (int r = 0; r < 8; ++r) {
        int Ml = r + 8 * hl;                            // pair row -> m within tile
        eg[(l_loc * 32 + mbase + Ml) * 16 + lm] = D[r] + bias_lane;
      }
    }
    __syncthreads();

    // ---- (b) scores: S = Q·K^T, scale, clip, +e, write H ----
    float Hv[2][8], gv[2][8];
#pragma unroll
    for (int mt = 0; mt < 2; ++mt) {
      const __bf16* krow = Kb + (((size_t)b * HEADS + h) * NN + m0 + mt * 16 + lm) * DOTD;
      FragB16 KB0, KB1;
      KB0.u[0] = *(const u32x4*)(krow + 8 * hl);
      KB0.u[1] = *(const u32x4*)(krow + 16 + 8 * hl);
      KB1.u[0] = *(const u32x4*)(krow + 32 + 8 * hl);
      KB1.u[1] = *(const u32x4*)(krow + 48 + 8 * hl);
      v8f S = wmma_bf16(QA[0].v, KB0.v, v8f_zero());
      S = wmma_bf16(QA[1].v, KB1.v, S);
#pragma unroll
      for (int r = 0; r < 8; ++r) {
        int Ml = r + 8 * hl;                            // l row within tile
        int m_loc = mt * 16 + lm;
        float s = S[r] * 0.125f;                        // 1/sqrt(64)
        s = fminf(fmaxf(s, -5.0f), 5.0f);
        float ev = eg[(Ml * 32 + m_loc) * 16 + h];
        float gg = eg[(Ml * 32 + m_loc) * 16 + 8 + h];
        float Hval = s + ev;
        H_out[((((size_t)b * NN + l0 + Ml) * NN) + m0 + m_loc) * HEADS + h] = Hval;
        Hv[mt][r] = Hval;
        gv[mt][r] = gg;
      }
    }

    // ---- (c) online softmax + sigmoid gate, stage P (bf16) ----
#pragma unroll
    for (int r = 0; r < 8; ++r) {
      float tm = fmaxf(Hv[0][r], Hv[1][r]);
#pragma unroll
      for (int mask = 1; mask < 16; mask <<= 1)
        tm = fmaxf(tm, __shfl_xor(tm, mask, 32));       // 16-lane row reduction
      float nm    = fmaxf(mrun[r], tm);
      float scale = __expf(mrun[r] - nm);
      float p0 = __expf(Hv[0][r] - nm);
      float p1 = __expf(Hv[1][r] - nm);
      float rs = p0 + p1;
#pragma unroll
      for (int mask = 1; mask < 16; mask <<= 1)
        rs += __shfl_xor(rs, mask, 32);
      lrun[r] = lrun[r] * scale + rs;
      mrun[r] = nm;
#pragma unroll
      for (int ct = 0; ct < 4; ++ct) O[ct][r] *= scale;
      float sg0 = 1.0f / (1.0f + __expf(-gv[0][r]));
      float sg1 = 1.0f / (1.0f + __expf(-gv[1][r]));
      int Ml = r + 8 * hl;
      Pst[wave][Ml * 32 + lm]      = (__bf16)(p0 * sg0);
      Pst[wave][Ml * 32 + 16 + lm] = (__bf16)(p1 * sg1);
    }

    // ---- (d) O += P · V  (wave-private LDS, DS ops in-order per wave) ----
    FragB16 PA;
    {
      const __bf16* prow = &Pst[wave][lm * 32];
      PA.u[0] = *(const u32x4*)(prow + 8 * hl);
      PA.u[1] = *(const u32x4*)(prow + 16 + 8 * hl);
    }
#pragma unroll
    for (int dt = 0; dt < 4; ++dt) {
      const __bf16* vrow = Vtb + (((size_t)b * HEADS + h) * DOTD + dt * 16 + lm) * NN +
                           m0;
      FragB16 VB;
      VB.u[0] = *(const u32x4*)(vrow + 8 * hl);
      VB.u[1] = *(const u32x4*)(vrow + 16 + 8 * hl);
      O[dt] = wmma_bf16(PA.v, VB.v, O[dt]);
    }
    __syncthreads();                                    // protect eg for next chunk
  }

  // ---- epilogue: divide by denominator, scatter attn ----
#pragma unroll
  for (int r = 0; r < 8; ++r) {
    float inv = 1.0f / lrun[r];
    int Ml = r + 8 * hl;
    size_t rowbase = ((size_t)b * NN + l0 + Ml) * NODE_WD;
#pragma unroll
    for (int dt = 0; dt < 4; ++dt) {
      int d = dt * 16 + lm;
      attn_out[rowbase + d * HEADS + h] = O[dt][r] * inv;   // col = d*8 + h
    }
  }
}

// ---------------------------------------------------------------------------
extern "C" void kernel_launch(void* const* d_in, const int* in_sizes, int n_in,
                              void* d_out, int out_size, void* d_ws, size_t ws_size,
                              hipStream_t stream) {
  (void)in_sizes; (void)n_in; (void)out_size; (void)ws_size;
  const float* nodes = (const float*)d_in[0];
  const float* edges = (const float*)d_in[1];
  const float* Wq = (const float*)d_in[2];
  const float* bq = (const float*)d_in[3];
  const float* Wk = (const float*)d_in[4];
  const float* bk = (const float*)d_in[5];
  const float* Wv = (const float*)d_in[6];
  const float* bv = (const float*)d_in[7];
  const float* Wg = (const float*)d_in[8];
  const float* bg = (const float*)d_in[9];
  const float* We = (const float*)d_in[10];
  const float* be = (const float*)d_in[11];

  char* ws = (char*)d_ws;                       // ~19 MB used
  __bf16* nodes_bf = (__bf16*)(ws);                                 // 4 MB
  __bf16* Wt       = (__bf16*)(ws + ((size_t)4  << 20));            // 1.5 MB
  __bf16* Wcombt   = (__bf16*)(ws + ((size_t)6  << 20));            // 1 KB
  __bf16* Qb       = (__bf16*)(ws + ((size_t)7  << 20));            // 4 MB
  __bf16* Kb       = (__bf16*)(ws + ((size_t)11 << 20));            // 4 MB
  __bf16* Vtb      = (__bf16*)(ws + ((size_t)15 << 20));            // 4 MB

  float* attn_out = (float*)d_out;                                  // [4,1024,512]
  float* H_out    = attn_out + (size_t)BSZ * NN * NODE_WD;          // [4,1024,1024,8]

  prep_kernel<<<8192, 256, 0, stream>>>(nodes, Wq, Wk, Wv, Wg, We,
                                        nodes_bf, Wt, Wcombt);
  qkv_kernel<<<768, 256, 0, stream>>>(nodes_bf, Wt, bq, bk, bv, Qb, Kb, Vtb);
  attn_kernel<<<256, 256, 0, stream>>>(edges, Qb, Kb, Vtb, Wcombt, bg, be,
                                       attn_out, H_out);
}
